// EdgePairEncoder_85383949845223
// MI455X (gfx1250) — matrix operand order repaired and checked by reading, not compiled, not run on hardware
//
#include <hip/hip_runtime.h>

typedef __attribute__((ext_vector_type(16))) _Float16 v16h;
typedef __attribute__((ext_vector_type(8)))  float    v8f;
typedef __attribute__((ext_vector_type(8)))  unsigned v8u;

#define DIST_BINS 24
#define NRES      256
#define NB        2
#define EMB       128
#define FAN       384            // 16 atom-pairs * 24 bins
#define RELK      32
#define A_STRIDE  386            // 384 + 2 pad (halves) -> conflict-free frag reads
#define KCHUNKS   12             // 384 / 32

__global__ __launch_bounds__(256)
void EdgePairEncoder_85383949845223_kernel(const float* __restrict__ x_pos,
                                           const float* __restrict__ rbf_w,
                                           const float* __restrict__ rbf_b,
                                           const float* __restrict__ relpos_w,
                                           const float* __restrict__ relpos_b,
                                           float* __restrict__ out)
{
    __shared__ float    s_dist[256];              // [r*16 + c1*4 + c2]
    __shared__ _Float16 s_A[16 * A_STRIDE];       // 16 rows x 384 f16 features

    const int t  = threadIdx.x;
    const int i  = blockIdx.x & (NRES - 1);       // residue i
    const int bb = blockIdx.x >> 8;               // batch

    const int wave = t >> 5;                      // 8 waves: emb slice wave*16
    const int lane = t & 31;
    const int hl   = lane >> 4;                   // half-wave select
    const int ln   = lane & 15;
    const int e    = (wave << 4) + ln;            // this lane's output column

    // ---- B fragments: rbf_w slice for this wave, loaded ONCE per block into
    //      registers (12 chunks x 8 VGPRs f16).  B 32x16, contiguous k order used here (A matches): lane = col N,
    //      lanes 0-15 hold K 0..15, lanes 16-31 hold K 16..31, pairs per VGPR.
    v16h bfrag[KCHUNKS];
    #pragma unroll
    for (int kk = 0; kk < KCHUNKS; ++kk) {
        v16h bf;
        #pragma unroll
        for (int v = 0; v < 8; ++v) {
            const int k0 = (kk << 5) + (hl << 4) + (v << 1);
            bf[2 * v]     = (_Float16)rbf_w[(size_t)k0 * EMB + e];
            bf[2 * v + 1] = (_Float16)rbf_w[(size_t)(k0 + 1) * EMB + e];
        }
        bfrag[kk] = bf;
    }

    const float bias = rbf_b[e] + relpos_b[e];    // per-column, j-invariant

    // per-thread distance-phase mapping (fixed across j-tiles)
    const int r  = t >> 4;
    const int c1 = (t >> 2) & 3;
    const int c2 = t & 3;
    const float* xi = x_pos + (((size_t)bb * NRES + i) * 4 + c1) * 3;
    const float xi0 = xi[0], xi1 = xi[1], xi2 = xi[2];

    // ---- Loop over 16 j-tiles of 16 residues, reusing register-resident B ----
    for (int jt = 0; jt < 16; ++jt) {
        const int j0 = jt << 4;

        __syncthreads();                          // previous tile fully consumed

        // Phase 1: 16x16 atom-pair distances
        {
            const float* xj = x_pos + (((size_t)bb * NRES + (j0 + r)) * 4 + c2) * 3;
            const float dx = xi0 - xj[0];
            const float dy = xi1 - xj[1];
            const float dz = xi2 - xj[2];
            s_dist[t] = sqrtf(dx * dx + dy * dy + dz * dz + 1e-12f);
        }
        __syncthreads();

        // Phase 2: Gaussian RBF expansion -> f16 A tile (16 x 384)
        #pragma unroll
        for (int k = 0; k < 24; ++k) {
            const int idx = t + (k << 8);         // 0..6143
            const int rr  = idx / FAN;
            const int f   = idx - rr * FAN;
            const int pp  = f / DIST_BINS;        // c1*4 + c2
            const int d   = f - pp * DIST_BINS;
            const float dist  = s_dist[(rr << 4) + pp];
            const float delta = dist - 0.5f * (float)d;
            s_A[rr * A_STRIDE + f] = (_Float16)__expf(-2.0f * delta * delta);
        }
        __syncthreads();

        // Phase 3: 12 WMMAs, A from LDS, B from registers
        v8f acc = {};
        #pragma unroll
        for (int kk = 0; kk < KCHUNKS; ++kk) {
            v8u au;
            #pragma unroll
            for (int v = 0; v < 8; ++v) {
                const int kloc = (hl << 4) + (v << 1);
                au[v] = *(const unsigned*)(s_A + ln * A_STRIDE + (kk << 5) + kloc);
            }
            acc = __builtin_amdgcn_wmma_f32_16x16x32_f16(
                      false, __builtin_bit_cast(v16h, au),
                      false, bfrag[kk], (short)0, acc, false, false);
        }

        // Epilogue: bias + relative-position embedding, store
        float* outp = out + ((((size_t)bb * NRES + i) * NRES) + j0) * EMB + e;
        #pragma unroll
        for (int v = 0; v < 8; ++v) {
            const int m = v + (hl << 3);          // output row 0..15 per C/D layout
            int rel = (j0 + m) - i;
            rel = rel < -RELK ? -RELK : (rel > RELK ? RELK : rel);
            outp[(size_t)m * EMB] = acc[v] + bias + relpos_w[(rel + RELK) * EMB + e];
        }
    }
}

extern "C" void kernel_launch(void* const* d_in, const int* in_sizes, int n_in,
                              void* d_out, int out_size, void* d_ws, size_t ws_size,
                              hipStream_t stream) {
    const float* x_pos    = (const float*)d_in[0];
    const float* rbf_w    = (const float*)d_in[1];
    const float* rbf_b    = (const float*)d_in[2];
    const float* relpos_w = (const float*)d_in[3];
    const float* relpos_b = (const float*)d_in[4];
    float* out = (float*)d_out;

    const int grid = NB * NRES;                   // 512 blocks: one (batch, i) each
    EdgePairEncoder_85383949845223_kernel<<<grid, 256, 0, stream>>>(
        x_pos, rbf_w, rbf_b, relpos_w, relpos_b, out);

    (void)in_sizes; (void)n_in; (void)out_size; (void)d_ws; (void)ws_size;
}
